// CorefModel_41300405518598
// MI455X (gfx1250) — compile-verified, matrix operand order checked by reading, not run on hardware
//
#include <hip/hip_runtime.h>
#include <hip/hip_bf16.h>
#include <math.h>

// Problem constants: B=8, W=512, M=128, P=50, V=50000, E=300, L=256, H=512

typedef __bf16 bf16;
typedef __attribute__((ext_vector_type(16))) __bf16 v16bf;
typedef __attribute__((ext_vector_type(8)))  float  v8f;

__device__ __forceinline__ bf16 f2bf(float f) {
  unsigned u = __builtin_bit_cast(unsigned, f);
  unsigned short r = (unsigned short)((u + 0x7FFFu + ((u >> 16) & 1u)) >> 16);
  return __builtin_bit_cast(bf16, r);
}

__device__ __forceinline__ float sigmoidf_(float x) { return 1.f / (1.f + __expf(-x)); }

// A fragment (16xK tile, row-major in LDS) per CDNA5 ISA 7.12.2.
__device__ __forceinline__ v16bf frag_a_lds(const bf16* tile, int ldk, int kb) {
  const int lane = threadIdx.x & 31;
  const int m = lane & 15, g = lane >> 4;
  const bf16* p = tile + m * ldk + kb + 8 * g;
  v16bf a;
#pragma unroll
  for (int i = 0; i < 8; ++i) { a[i] = p[i]; a[i + 8] = p[i + 16]; }
  return a;
}

// B fragments pre-packed: frag = 32 lanes x 16 bf16 contiguous (32B per lane)
__device__ __forceinline__ v16bf frag_b_packed(const bf16* packed, int fragid) {
  const bf16* p = packed + ((size_t)fragid * 32 + (threadIdx.x & 31)) * 16;
  return *(const v16bf*)p;
}

__device__ __forceinline__ v8f wmma_bf16(v16bf a, v16bf b, v8f c) {
  return __builtin_amdgcn_wmma_f32_16x16x32_bf16(false, a, false, b, (short)0, c, false, false);
}

// CDNA5 async global->LDS copy (ASYNCcnt-tracked). LDS offset = low 32 bits of
// the generic shared pointer (ISA flat aperture rule).
__device__ __forceinline__ void async_load_lds_b128(void* lds_dst, const void* gsrc) {
  unsigned ldsoff = (unsigned)(size_t)lds_dst;
  asm volatile("global_load_async_to_lds_b128 %0, %1, off"
               :: "v"(ldsoff), "v"(gsrc) : "memory");
}
__device__ __forceinline__ void wait_asynccnt0() {
  asm volatile("s_wait_asynccnt 0x0" ::: "memory");
}

// ---------------------------------------------------------------------------
// Pack f32 KxN row-major weight into per-fragment B layout (bf16), zero-pad K.
__global__ void pack_b_kernel(const float* __restrict__ src, bf16* __restrict__ dst,
                              int K, int N, int Ksrc) {
  const int idx = blockIdx.x * blockDim.x + threadIdx.x;
  const int nT = N / 16;
  const int total = (K / 32) * nT * 32;
  if (idx >= total) return;
  const int lane = idx & 31;
  const int fragid = idx >> 5;
  const int nt = fragid % nT, kt = fragid / nT;
  const int n = nt * 16 + (lane & 15);
  const int g = lane >> 4;
  bf16* out = dst + (size_t)fragid * 512 + lane * 16;
#pragma unroll
  for (int i = 0; i < 16; ++i) {
    const int kl = (i < 8) ? (8 * g + i) : (16 + 8 * g + (i - 8));
    const int k = kt * 32 + kl;
    const float v = (k < Ksrc) ? src[(size_t)k * N + n] : 0.f;
    out[i] = f2bf(v);
  }
}

// ---------------------------------------------------------------------------
// Xbf[b*512+t][1024] = bf16( emb[word_seq] @ Wih + bih + bhh ). 2 waves/block.
__global__ void embed_gemm_kernel(const int* __restrict__ word_seq,
                                  const float* __restrict__ emb,
                                  const bf16* __restrict__ WihP,
                                  const float* __restrict__ bih,
                                  const float* __restrict__ bhh,
                                  bf16* __restrict__ Xbf) {
  __shared__ bf16 Atile[2][16 * 320];
  const int wave = threadIdx.x >> 5, lane = threadIdx.x & 31;
  const int tile = blockIdx.x * 2 + wave;
  bf16* A = Atile[wave];
  for (int r = 0; r < 16; ++r) {
    const int row = tile * 16 + r;
    const float* e = emb + (size_t)word_seq[row] * 300;
    for (int u = lane; u < 320; u += 32)
      A[r * 320 + u] = (u < 300) ? f2bf(e[u]) : f2bf(0.f);
  }
  __syncthreads();
  const int n15 = lane & 15, g = lane >> 4;
  for (int nt = 0; nt < 64; ++nt) {
    v8f acc = {};
#pragma unroll
    for (int kt = 0; kt < 10; ++kt)
      acc = wmma_bf16(frag_a_lds(A, 320, kt * 32), frag_b_packed(WihP, kt * 64 + nt), acc);
    const int n = nt * 16 + n15;
    const float bias = bih[n] + bhh[n];
#pragma unroll
    for (int i = 0; i < 8; ++i)
      Xbf[(size_t)(tile * 16 + i + 8 * g) * 1024 + n] = f2bf(acc[i] + bias);
  }
}

// ---------------------------------------------------------------------------
// Persistent LSTM: single workgroup (32 waves), 512 serial steps.
// X slab for step t is async-copied to LDS while the WMMA recurrence for
// step t runs (ASYNCcnt pipelining). LDS total: 56 KB.
__global__ void lstm_kernel(const bf16* __restrict__ Xbf,
                            const bf16* __restrict__ WhhP,
                            float* __restrict__ csum) {
  __shared__ bf16 hbf[16 * 256];      // rows 8..15 stay zero (8 KB)
  __shared__ bf16 G[8 * 1024];        // gate pre-activations (16 KB)
  __shared__ float cst[8 * 256];      // cell state (8 KB)
  __shared__ float csacc[8 * 256];    // running cumsum (8 KB)
  __shared__ bf16 Xb[8 * 1024];       // async-staged X slab (16 KB)
  const int tid = threadIdx.x;        // 1024 threads
  const int lane = tid & 31, wave = tid >> 5;
  for (int idx = tid; idx < 16 * 256; idx += 1024) hbf[idx] = f2bf(0.f);
  for (int idx = tid; idx < 8 * 256; idx += 1024) {
    cst[idx] = 0.f; csacc[idx] = 0.f;
    const int b = idx >> 8, u = idx & 255;
    csum[(size_t)(b * 513) * 256 + u] = 0.f;
  }
  // issue async copy of X(t=0): 1024 threads x one b128 chunk (8 bf16)
  {
    const int b = tid >> 7, o8 = (tid & 127) * 8;
    async_load_lds_b128(&Xb[b * 1024 + o8], Xbf + (size_t)(b * 512 + 0) * 1024 + o8);
  }
  __syncthreads();
  for (int t = 0; t < 512; ++t) {
    // G = h @ Whh : 2 N-tiles per wave, overlaps in-flight async X copy
#pragma unroll
    for (int q = 0; q < 2; ++q) {
      const int nt = wave * 2 + q;
      v8f acc = {};
#pragma unroll
      for (int kt = 0; kt < 8; ++kt)
        acc = wmma_bf16(frag_a_lds(hbf, 256, kt * 32), frag_b_packed(WhhP, kt * 64 + nt), acc);
      if (lane < 16) {                 // rows 0..7 only (B=8)
#pragma unroll
        for (int i = 0; i < 8; ++i) G[i * 1024 + nt * 16 + lane] = f2bf(acc[i]);
      }
    }
    wait_asynccnt0();                  // this wave's X(t) chunks have landed
    __syncthreads();                   // all waves: G + Xb(t) visible
#pragma unroll
    for (int q = 0; q < 2; ++q) {
      const int idx = tid + q * 1024;  // 0..2047 = b*256+u
      const int b = idx >> 8, u = idx & 255;
      const bf16* xr = Xb + b * 1024;
      const float gi = (float)G[b * 1024 + u]       + (float)xr[u];
      const float gf = (float)G[b * 1024 + 256 + u] + (float)xr[256 + u];
      const float gg = (float)G[b * 1024 + 512 + u] + (float)xr[512 + u];
      const float go = (float)G[b * 1024 + 768 + u] + (float)xr[768 + u];
      const float cn = sigmoidf_(gf) * cst[idx] + sigmoidf_(gi) * tanhf(gg);
      const float hn = sigmoidf_(go) * tanhf(cn);
      cst[idx] = cn;
      const float cs = csacc[idx] + hn;
      csacc[idx] = cs;
      csum[(size_t)(b * 513 + t + 1) * 256 + u] = cs;
      hbf[idx] = f2bf(hn);
    }
    __syncthreads();                   // gates done reading Xb
    if (t + 1 < 512) {                 // prefetch X(t+1); overlaps next WMMA phase
      const int b = tid >> 7, o8 = (tid & 127) * 8;
      async_load_lds_b128(&Xb[b * 1024 + o8],
                          Xbf + (size_t)(b * 512 + t + 1) * 1024 + o8);
    }
  }
}

// ---------------------------------------------------------------------------
// tgt[b*128+i][256] = csum[b][end]-csum[b][start]
__global__ void span_kernel(const float* __restrict__ csum,
                            const int* __restrict__ starts,
                            const int* __restrict__ lens,
                            float* __restrict__ tgt) {
  const int idx = blockIdx.x * blockDim.x + threadIdx.x;
  if (idx >= 1024 * 256) return;
  const int u = idx & 255, row = idx >> 8;
  const int b = row >> 7, i = row & 127;
  int s = starts[b * 128 + i];
  int e = s + lens[b * 128 + i];
  e = e < 0 ? 0 : (e > 512 ? 512 : e);
  s = s < 0 ? 0 : (s > 512 ? 512 : s);
  tgt[(size_t)row * 256 + u] =
      csum[(size_t)(b * 513 + e) * 256 + u] - csum[(size_t)(b * 513 + s) * 256 + u];
}

// ---------------------------------------------------------------------------
// Mention MLP: 64 blocks (1 M-tile each, 8 waves)
__global__ void mention_kernel(const float* __restrict__ tgt,
                               const bf16* __restrict__ Wm1P, const float* __restrict__ bm1,
                               const bf16* __restrict__ Wm2P, const float* __restrict__ bm2,
                               const float* __restrict__ wm,
                               float* __restrict__ ms) {
  __shared__ bf16 A1[16 * 256];
  __shared__ bf16 H1[16 * 512];
  __shared__ float H2[16 * 512];
  __shared__ float part[16 * 16];
  const int tid = threadIdx.x;         // 256 threads
  const int lane = tid & 31, wave = tid >> 5;
  const int tile = blockIdx.x;
  for (int idx = tid; idx < 16 * 256; idx += 256)
    A1[idx] = f2bf(tgt[(size_t)tile * 16 * 256 + idx]);
  __syncthreads();
  const int n15 = lane & 15, g = lane >> 4;
  for (int q = 0; q < 4; ++q) {
    const int nt = wave * 4 + q;
    v8f acc = {};
#pragma unroll
    for (int kt = 0; kt < 8; ++kt)
      acc = wmma_bf16(frag_a_lds(A1, 256, kt * 32), frag_b_packed(Wm1P, kt * 32 + nt), acc);
    const int n = nt * 16 + n15;
    const float b1 = bm1[n];
#pragma unroll
    for (int i = 0; i < 8; ++i)
      H1[(i + 8 * g) * 512 + n] = f2bf(fmaxf(acc[i] + b1, 0.f));
  }
  __syncthreads();
  for (int q = 0; q < 4; ++q) {
    const int nt = wave * 4 + q;
    v8f acc = {};
#pragma unroll
    for (int kt = 0; kt < 16; ++kt)
      acc = wmma_bf16(frag_a_lds(H1, 512, kt * 32), frag_b_packed(Wm2P, kt * 32 + nt), acc);
    const int n = nt * 16 + n15;
    const float b2 = bm2[n];
#pragma unroll
    for (int i = 0; i < 8; ++i)
      H2[(i + 8 * g) * 512 + n] = fmaxf(acc[i] + b2, 0.f);
  }
  __syncthreads();
  {
    const int r = tid >> 4, seg = tid & 15;
    float s = 0.f;
    for (int u = seg * 32; u < seg * 32 + 32; ++u) s += H2[r * 512 + u] * wm[u];
    part[r * 16 + seg] = s;
  }
  __syncthreads();
  if (tid < 16) {
    float s = 0.f;
#pragma unroll
    for (int j = 0; j < 16; ++j) s += part[tid * 16 + j];
    ms[tile * 16 + tid] = s;
  }
}

// ---------------------------------------------------------------------------
// Pair MLP (dominant ~67 GFLOP): 3200 blocks x 128 threads (4 waves share one
// 16-row tile: 40 KB LDS per 4 waves -> 8 blocks/WGP occupancy). Each wave
// owns 8 of 32 N-tiles. feat never materialized in HBM.
__global__ void pair_kernel(const float* __restrict__ tgt,
                            const bf16* __restrict__ Wa1P, const float* __restrict__ ba1,
                            const bf16* __restrict__ Wa2P, const float* __restrict__ ba2,
                            const float* __restrict__ wa,
                            const float* __restrict__ ms,
                            float* __restrict__ scores) {
  __shared__ bf16 fA[16 * 768];
  __shared__ bf16 H1[16 * 512];
  __shared__ float ps[16];
  const int tid = threadIdx.x;         // 128 threads = 4 waves
  const int lane = tid & 31, wave = tid >> 5;
  const int tile = blockIdx.x;
  // cooperative feature-tile build: [jvec | ivec | jvec*ivec]
  for (int idx = tid; idx < 16 * 256; idx += 128) {
    const int r = idx >> 8, u = idx & 255;
    const int row = tile * 16 + r;
    const int b = row / 6400, rem = row % 6400;
    const int mi = rem / 50, k = rem % 50;
    int j = (mi - 50 > 0 ? mi - 50 : 0) + k;
    const int jc = j > 127 ? 127 : j;
    const float a = tgt[(size_t)(b * 128 + jc) * 256 + u];
    const float c = tgt[(size_t)(b * 128 + mi) * 256 + u];
    fA[r * 768 + u]       = f2bf(a);
    fA[r * 768 + 256 + u] = f2bf(c);
    fA[r * 768 + 512 + u] = f2bf(a * c);
  }
  if (tid < 16) ps[tid] = 0.f;
  __syncthreads();
  const int n15 = lane & 15, g = lane >> 4;
  for (int q = 0; q < 8; ++q) {
    const int nt = wave * 8 + q;
    v8f acc = {};
#pragma unroll
    for (int kt = 0; kt < 24; ++kt)
      acc = wmma_bf16(frag_a_lds(fA, 768, kt * 32), frag_b_packed(Wa1P, kt * 32 + nt), acc);
    const int n = nt * 16 + n15;
    const float b1 = ba1[n];
#pragma unroll
    for (int i = 0; i < 8; ++i)
      H1[(i + 8 * g) * 512 + n] = f2bf(fmaxf(acc[i] + b1, 0.f));
  }
  __syncthreads();
  for (int q = 0; q < 8; ++q) {
    const int nt = wave * 8 + q;
    v8f acc = {};
#pragma unroll
    for (int kt = 0; kt < 16; ++kt)
      acc = wmma_bf16(frag_a_lds(H1, 512, kt * 32), frag_b_packed(Wa2P, kt * 32 + nt), acc);
    const int n = nt * 16 + n15;
    const float b2 = ba2[n];
    const float wn = wa[n];
#pragma unroll
    for (int i = 0; i < 8; ++i) {
      const float v = fmaxf(acc[i] + b2, 0.f);
      atomicAdd(&ps[i + 8 * g], v * wn);   // ds_add_f32
    }
  }
  __syncthreads();
  if (tid < 16) {
    const int row = tile * 16 + tid;
    const int b = row / 6400, rem = row % 6400;
    const int mi = rem / 50, k = rem % 50;
    int j = (mi - 50 > 0 ? mi - 50 : 0) + k;
    const int jc = j > 127 ? 127 : j;
    const int lim = mi < 50 ? mi : 50;
    scores[row] = (k < lim) ? (ms[b * 128 + mi] + ms[b * 128 + jc] + ps[tid])
                            : -INFINITY;
  }
}

// ---------------------------------------------------------------------------
// Softmax over 50 scores + appended 0 -> out[b*128+i][51]
__global__ void softmax_kernel(const float* __restrict__ scores, float* __restrict__ out) {
  __shared__ float red[64];
  __shared__ float vals[51];
  const int row = blockIdx.x;          // 0..1023
  const int t = threadIdx.x;           // 64 threads
  float v = -INFINITY;
  if (t < 51) { v = (t < 50) ? scores[(size_t)row * 50 + t] : 0.f; vals[t] = v; }
  red[t] = v;
  __syncthreads();
  for (int s = 32; s > 0; s >>= 1) {
    if (t < s) red[t] = fmaxf(red[t], red[t + s]);
    __syncthreads();
  }
  const float mx = red[0];
  __syncthreads();
  float e = 0.f;
  if (t < 51) e = expf(vals[t] - mx);
  red[t] = e;
  __syncthreads();
  for (int s = 32; s > 0; s >>= 1) {
    if (t < s) red[t] += red[t + s];
    __syncthreads();
  }
  const float sum = red[0];
  if (t < 51) out[(size_t)row * 51 + t] = e / sum;
}

// ---------------------------------------------------------------------------
extern "C" void kernel_launch(void* const* d_in, const int* in_sizes, int n_in,
                              void* d_out, int out_size, void* d_ws, size_t ws_size,
                              hipStream_t stream) {
  (void)in_sizes; (void)n_in; (void)out_size; (void)ws_size;
  const int*   word_seq = (const int*)d_in[0];
  const int*   starts   = (const int*)d_in[1];
  const int*   lens     = (const int*)d_in[2];
  const float* emb = (const float*)d_in[3];
  const float* Wih = (const float*)d_in[4];
  const float* Whh = (const float*)d_in[5];
  const float* bih = (const float*)d_in[6];
  const float* bhh = (const float*)d_in[7];
  const float* Wm1 = (const float*)d_in[8];
  const float* bm1 = (const float*)d_in[9];
  const float* Wm2 = (const float*)d_in[10];
  const float* bm2 = (const float*)d_in[11];
  const float* wm  = (const float*)d_in[12];
  const float* Wa1 = (const float*)d_in[13];
  const float* ba1 = (const float*)d_in[14];
  const float* Wa2 = (const float*)d_in[15];
  const float* ba2 = (const float*)d_in[16];
  const float* wa  = (const float*)d_in[17];

  char* ws = (char*)d_ws;
  size_t off = 0;
  auto take = [&](size_t bytes) {
    void* p = ws + off;
    off = (off + bytes + 255) & ~(size_t)255;
    return p;
  };
  bf16*  Xbf  = (bf16*)take(4096ull * 1024 * 2);
  float* csum = (float*)take(8ull * 513 * 256 * 4);
  float* tgt  = (float*)take(1024ull * 256 * 4);
  float* ms   = (float*)take(1024 * 4);
  float* sc   = (float*)take(51200 * 4);
  bf16* WihP = (bf16*)take(640ull * 512 * 2);  // 10x64 frags
  bf16* WhhP = (bf16*)take(512ull * 512 * 2);  // 8x64
  bf16* Wm1P = (bf16*)take(256ull * 512 * 2);  // 8x32
  bf16* Wm2P = (bf16*)take(512ull * 512 * 2);  // 16x32
  bf16* Wa1P = (bf16*)take(768ull * 512 * 2);  // 24x32
  bf16* Wa2P = (bf16*)take(512ull * 512 * 2);  // 16x32

  pack_b_kernel<<<80, 256, 0, stream>>>(Wih, WihP, 320, 1024, 300);
  pack_b_kernel<<<64, 256, 0, stream>>>(Whh, WhhP, 256, 1024, 256);
  pack_b_kernel<<<32, 256, 0, stream>>>(Wm1, Wm1P, 256, 512, 256);
  pack_b_kernel<<<64, 256, 0, stream>>>(Wm2, Wm2P, 512, 512, 512);
  pack_b_kernel<<<96, 256, 0, stream>>>(Wa1, Wa1P, 768, 512, 768);
  pack_b_kernel<<<64, 256, 0, stream>>>(Wa2, Wa2P, 512, 512, 512);

  embed_gemm_kernel<<<128, 64, 0, stream>>>(word_seq, emb, WihP, bih, bhh, Xbf);
  lstm_kernel<<<1, 1024, 0, stream>>>(Xbf, WhhP, csum);
  span_kernel<<<1024, 256, 0, stream>>>(csum, starts, lens, tgt);
  mention_kernel<<<64, 256, 0, stream>>>(tgt, Wm1P, bm1, Wm2P, bm2, wm, ms);
  pair_kernel<<<3200, 128, 0, stream>>>(tgt, Wa1P, ba1, Wa2P, ba2, wa, ms, sc);
  softmax_kernel<<<1024, 64, 0, stream>>>(sc, (float*)d_out);
}
